// InteractionNetwork_43087111914204
// MI455X (gfx1250) — matrix operand order repaired and checked by reading, not compiled, not run on hardware
//
#include <hip/hip_runtime.h>
#include <hip/hip_bf16.h>
#include <math.h>

// ---------------------------------------------------------------------------
// Interaction Network on MI455X (gfx1250), bf16 WMMA everywhere.
// Pipeline: pack weights -> gather (WMMA) -> 4x MLP layers (WMMA, LDS-staged
// weights) -> scatter (WMMA + f32 atomics) -> object MLP (WMMA + shuffle
// reduce + sigmoid). Intermediates stay L2-resident (192MB L2).
// Round 3: dual WMMA accumulators (break the D->C serial chain so the matrix
// pipe issues back-to-back); vectorized relation_info passthrough.
// ---------------------------------------------------------------------------

typedef __attribute__((ext_vector_type(16))) __bf16 v16bf;
typedef __attribute__((ext_vector_type(8)))  __bf16 v8bf;
typedef __attribute__((ext_vector_type(8)))  float  v8f;
typedef __attribute__((ext_vector_type(4)))  float  v4f;

constexpr int  Bn    = 16;
constexpr int  Nn    = 128;
constexpr int  Rn    = 16256;
constexpr int  OBJ_D = 64;
constexpr int  REL_D = 32;
constexpr int  XK    = 160;   // 64 + 64 + 32 input features
constexpr int  HP    = 160;   // REL_HID 150 padded to 160
constexpr int  EP    = 112;   // EFF_D 100 padded to 112
constexpr long MROWS = (long)Bn * Rn;   // 260096 relation rows

// K-offset inside a 32-wide chunk for the 16-bit WMMA A layout:
// lanes 0-15 hold K = {0..7, 16..23}, lanes 16-31 hold K = {8..15, 24..31}.
__device__ __forceinline__ int kidx(int e, int half) {
  return (e < 8) ? (half * 8 + e) : (16 + half * 8 + (e - 8));
}

__device__ __forceinline__ v8f wmma_bf16(v16bf a, v16bf b, v8f c) {
  return __builtin_amdgcn_wmma_f32_16x16x32_bf16(
      /*neg_a=*/false, a, /*neg_b=*/false, b,
      /*c_mod=*/(short)0, c, /*reuse_a=*/false, /*reuse_b=*/false);
}

// ---------------------------------------------------------------------------
// Prep kernels
// ---------------------------------------------------------------------------

// Pack a row-major f32 weight [Kreal x Nreal] into per-lane bf16 B-fragments:
// dst[((kt*NCT+ct)*32 + lane)*16 + e] = W[kt*32 + kidx(e, lane>>4)][ct*16 + (lane&15)]
__global__ void pack_weight_kernel(const float* __restrict__ src,
                                   __bf16* __restrict__ dst,
                                   int Kreal, int Nreal, int NKT, int NCT) {
  int idx = blockIdx.x * blockDim.x + threadIdx.x;
  int total = NKT * NCT * 512;
  if (idx >= total) return;
  int e    = idx & 15;
  int lane = (idx >> 4) & 31;
  int rem  = idx >> 9;
  int ct   = rem % NCT;
  int kt   = rem / NCT;
  int k = kt * 32 + kidx(e, lane >> 4);
  int n = ct * 16 + (lane & 15);
  float v = (k < Kreal && n < Nreal) ? src[(size_t)k * Nreal + n] : 0.0f;
  dst[idx] = (__bf16)v;
}

// Pack objects [B,128,64] f32 into per-batch bf16 B-fragments (K=n, N=d).
__global__ void pack_objects_kernel(const float* __restrict__ objs,
                                    __bf16* __restrict__ dst) {
  int idx = blockIdx.x * blockDim.x + threadIdx.x;
  if (idx >= Bn * 16 * 512) return;
  int e    = idx & 15;
  int lane = (idx >> 4) & 31;
  int rem  = idx >> 9;
  int ct   = rem % 4;
  int kt   = (rem / 4) % 4;
  int b    = rem / 16;
  int k   = kt * 32 + kidx(e, lane >> 4);    // object index n: 0..127
  int col = ct * 16 + (lane & 15);           // feature d:     0..63
  dst[idx] = (__bf16)objs[((size_t)b * Nn + k) * OBJ_D + col];
}

// Pad the 5 bias vectors to 160 f32 slots each (zero pad).
__global__ void pack_bias_kernel(const float* __restrict__ b1,
                                 const float* __restrict__ b2,
                                 const float* __restrict__ b3,
                                 const float* __restrict__ b4,
                                 const float* __restrict__ b5,
                                 float* __restrict__ dst) {
  int idx = blockIdx.x * blockDim.x + threadIdx.x;
  if (idx >= 5 * 160) return;
  int which = idx / 160, j = idx % 160;
  const float* s = b1; int nr = 150;
  if (which == 1) { s = b2; nr = 150; }
  else if (which == 2) { s = b3; nr = 150; }
  else if (which == 3) { s = b4; nr = 100; }
  else if (which == 4) { s = b5; nr = 100; }
  dst[idx] = (j < nr) ? s[j] : 0.0f;
}

__global__ void zero_kernel(float* __restrict__ p, long n) {
  long i = (long)blockIdx.x * blockDim.x + threadIdx.x;
  if (i < n) p[i] = 0.0f;
}

// ---------------------------------------------------------------------------
// Gather: X[b*R+r, 0:64]   = (S^T O)[r,:]   (senders)
//         X[b*R+r, 64:128] = (R^T O)[r,:]   (receivers)
//         X[b*R+r, 128:160]= relation_info[b,r,:]
// One wave per 16-row r-tile; objects B-fragments staged in LDS per batch.
// ---------------------------------------------------------------------------
__global__ void gather_kernel(const float* __restrict__ srel,
                              const float* __restrict__ rrel,
                              const float* __restrict__ rinfo,
                              const __bf16* __restrict__ objp,
                              __bf16* __restrict__ X) {
  __shared__ __align__(32) __bf16 sObj[4 * 4 * 512];  // 16KB
  int tid = threadIdx.x;
  int bx  = blockIdx.x;
  int b   = bx / 254;
  int tg  = bx % 254;

  {  // stage packed objects for this batch
    const uint32_t* s = (const uint32_t*)(objp + (size_t)b * 8192);
    uint32_t* d = (uint32_t*)sObj;
    for (int i = tid; i < 4096; i += blockDim.x) d[i] = s[i];
  }
  __syncthreads();

  int w = tid / 32, lane = tid % 32;
  int half = lane >> 4, lm = lane & 15;
  int rtile = tg * 4 + w;            // 0..1015
  int r0    = rtile * 16;
  int row_r = r0 + lm;

#pragma unroll 1
  for (int s = 0; s < 2; ++s) {
    const float* rel = s ? rrel : srel;
    int coloff = s * OBJ_D;
    v16bf aF[4];
#pragma unroll
    for (int kt = 0; kt < 4; ++kt) {
#pragma unroll
      for (int e = 0; e < 16; ++e) {
        int n = kt * 32 + kidx(e, half);
        aF[kt][e] = (__bf16)rel[((size_t)b * Nn + n) * Rn + row_r];
      }
    }
    __builtin_prefetch(&rel[((size_t)b * Nn) * Rn + row_r + 16], 0, 3);
#pragma unroll 1
    for (int ct = 0; ct < 4; ++ct) {
      // dual accumulators: break the WMMA D->C dependency chain
      v8f acc0 = {}, acc1 = {};
      {
        v16bf b0 = *(const v16bf*)&sObj[((0 * 4 + ct) * 32 + lane) * 16];
        v16bf b1 = *(const v16bf*)&sObj[((1 * 4 + ct) * 32 + lane) * 16];
        v16bf b2 = *(const v16bf*)&sObj[((2 * 4 + ct) * 32 + lane) * 16];
        v16bf b3 = *(const v16bf*)&sObj[((3 * 4 + ct) * 32 + lane) * 16];
        acc0 = wmma_bf16(aF[0], b0, acc0);
        acc1 = wmma_bf16(aF[1], b1, acc1);
        acc0 = wmma_bf16(aF[2], b2, acc0);
        acc1 = wmma_bf16(aF[3], b3, acc1);
      }
#pragma unroll
      for (int v = 0; v < 8; ++v) {
        int rr = r0 + v + 8 * half;
        X[((size_t)b * Rn + rr) * XK + coloff + ct * 16 + lm] =
            (__bf16)(acc0[v] + acc1[v]);
      }
    }
  }

  // relation_info passthrough, cols 128..159 (16 contiguous bf16 per lane)
  {
    int rrow = r0 + (lane >> 1);
    int c0   = (lane & 1) * 16;
    const float* rsrc = rinfo + ((size_t)b * Rn + rrow) * REL_D + c0;
    v4f p0 = *(const v4f*)(rsrc + 0);
    v4f p1 = *(const v4f*)(rsrc + 4);
    v4f p2 = *(const v4f*)(rsrc + 8);
    v4f p3 = *(const v4f*)(rsrc + 12);
    v16bf o;
#pragma unroll
    for (int i = 0; i < 4; ++i) {
      o[i]      = (__bf16)p0[i];
      o[4 + i]  = (__bf16)p1[i];
      o[8 + i]  = (__bf16)p2[i];
      o[12 + i] = (__bf16)p3[i];
    }
    *(v16bf*)&X[((size_t)b * Rn + rrow) * XK + 128 + c0] = o;
  }
}

// ---------------------------------------------------------------------------
// Dense layer: out = relu(act @ W + bias), bf16 in/out, f32 accumulate.
// 8 waves per WG, one 16-row tile per wave, packed weights staged in LDS.
// ct loop deliberately NOT unrolled (prevents hoist-and-spill of B frags).
// ---------------------------------------------------------------------------
template <int NKT, int NCT>
__global__ void layer_kernel(const __bf16* __restrict__ act, int istride,
                             __bf16* __restrict__ out, int ostride,
                             const __bf16* __restrict__ Wp,
                             const float* __restrict__ bias) {
  __shared__ __align__(32) __bf16 sW[NKT * NCT * 512];
  __shared__ float sB[160];
  int tid = threadIdx.x;
  {
    const uint32_t* s = (const uint32_t*)Wp;
    uint32_t* d = (uint32_t*)sW;
    for (int i = tid; i < NKT * NCT * 256; i += blockDim.x) d[i] = s[i];
    for (int i = tid; i < 160; i += blockDim.x) sB[i] = bias[i];
  }
  __syncthreads();

  int w = tid / 32, lane = tid % 32;
  int half = lane >> 4, lm = lane & 15;
  long rtile = (long)blockIdx.x * 8 + w;
  long r0    = rtile * 16;

  const __bf16* arow = act + (r0 + lm) * istride;
  v16bf aF[NKT];
#pragma unroll
  for (int kt = 0; kt < NKT; ++kt) {
    v8bf lo = *(const v8bf*)(arow + kt * 32 + half * 8);
    v8bf hi = *(const v8bf*)(arow + kt * 32 + 16 + half * 8);
#pragma unroll
    for (int i = 0; i < 8; ++i) { aF[kt][i] = lo[i]; aF[kt][8 + i] = hi[i]; }
  }

#pragma unroll 1
  for (int ct = 0; ct < NCT; ++ct) {
    // dual accumulators over the kt chain
    v8f acc0 = {}, acc1 = {};
#pragma unroll
    for (int kt = 0; kt < NKT; ++kt) {
      v16bf bF = *(const v16bf*)&sW[((kt * NCT + ct) * 32 + lane) * 16];
      if (kt & 1) acc1 = wmma_bf16(aF[kt], bF, acc1);
      else        acc0 = wmma_bf16(aF[kt], bF, acc0);
    }
    float bv = sB[ct * 16 + lm];
    __bf16* orow = out + (r0 + 8 * half) * ostride + ct * 16 + lm;
#pragma unroll
    for (int v = 0; v < 8; ++v) {
      float r = acc0[v] + acc1[v] + bv;
      orow[(size_t)v * ostride] = (__bf16)(r > 0.0f ? r : 0.0f);
    }
  }
}

// ---------------------------------------------------------------------------
// Scatter: ER[b,n,:] = sum_r recv[b,n,r] * effects[b*R+r,:]
// K=16256 split into 8 chunks per (b, n-tile, c-tile) wave; f32 atomics.
// A-fragment lanes hold 8+8 consecutive r values -> vector float4 loads.
// K loop unrolled x2 with alternating accumulators.
// ---------------------------------------------------------------------------
__global__ void scatter_kernel(const float* __restrict__ rrel,
                               const __bf16* __restrict__ eff,
                               float* __restrict__ er) {
  int tid = threadIdx.x;
  int lane = tid % 32, w = tid / 32;
  int half = lane >> 4, lm = lane & 15;
  int g  = blockIdx.x * 8 + w;
  int kc = g % 8;  g /= 8;
  int ct = g % 7;  g /= 7;
  int nt = g % 8;
  int b  = g / 8;

  const int KSTEPS = Rn / 32;  // 508
  int k0 = kc * 64;
  int k1 = (k0 + 64 < KSTEPS) ? (k0 + 64) : KSTEPS;  // 64 or 60 steps (even)
  int n  = nt * 16 + lm;
  const float* arow = rrel + ((size_t)b * Nn + n) * Rn;

  v8f acc0 = {}, acc1 = {};
#pragma unroll 1
  for (int ks = k0; ks < k1; ks += 2) {
#pragma unroll
    for (int u = 0; u < 2; ++u) {
      int rbase = (ks + u) * 32;
      // A: elems 0-7 = r [rbase+half*8,+8), elems 8-15 = r [rbase+16+half*8,+8)
      v4f q0 = *(const v4f*)(arow + rbase + half * 8);
      v4f q1 = *(const v4f*)(arow + rbase + half * 8 + 4);
      v4f q2 = *(const v4f*)(arow + rbase + 16 + half * 8);
      v4f q3 = *(const v4f*)(arow + rbase + 16 + half * 8 + 4);
      v16bf aF, bF;
#pragma unroll
      for (int i = 0; i < 4; ++i) {
        aF[i]      = (__bf16)q0[i];
        aF[4 + i]  = (__bf16)q1[i];
        aF[8 + i]  = (__bf16)q2[i];
        aF[12 + i] = (__bf16)q3[i];
      }
#pragma unroll
      for (int e = 0; e < 16; ++e) {
        int r = rbase + kidx(e, half);
        bF[e] = eff[((size_t)b * Rn + r) * EP + ct * 16 + lm];
      }
      if (u) acc1 = wmma_bf16(aF, bF, acc1);
      else   acc0 = wmma_bf16(aF, bF, acc0);
    }
    __builtin_prefetch(arow + (ks + 2) * 32, 0, 3);
  }
#pragma unroll
  for (int v = 0; v < 8; ++v) {
    atomicAdd(&er[((size_t)b * Nn + nt * 16 + v + 8 * half) * EP + ct * 16 + lm],
              acc0[v] + acc1[v]);
  }
}

// ---------------------------------------------------------------------------
// Object MLP: y=[objects | effect_receivers] (164 -> pad 192)
// h = relu(y @ ow1 + ob1) via WMMA, then per-row dot with ow2 via shuffle
// reduction, + sigmoid.
// ---------------------------------------------------------------------------
__global__ void final_kernel(const float* __restrict__ objects,
                             const float* __restrict__ er,
                             const __bf16* __restrict__ ow1p,
                             const float* __restrict__ ob1p,
                             const float* __restrict__ ow2,
                             const float* __restrict__ ob2,
                             float* __restrict__ out) {
  __shared__ __align__(32) __bf16 sW[6 * 7 * 512];  // 43KB
  __shared__ float sB[160];
  int tid = threadIdx.x;
  {
    const uint32_t* s = (const uint32_t*)ow1p;
    uint32_t* d = (uint32_t*)sW;
    for (int i = tid; i < 6 * 7 * 256; i += blockDim.x) d[i] = s[i];
    for (int i = tid; i < 160; i += blockDim.x) sB[i] = ob1p[i];
  }
  __syncthreads();

  int w = tid / 32, lane = tid % 32;
  int half = lane >> 4, lm = lane & 15;
  int rtile = blockIdx.x * 4 + w;
  int r0    = rtile * 16;
  int orow  = r0 + lm;

  v16bf aF[6];
#pragma unroll
  for (int kt = 0; kt < 6; ++kt) {
#pragma unroll
    for (int e = 0; e < 16; ++e) {
      int k = kt * 32 + kidx(e, half);
      float v = 0.0f;
      if (k < 64)        v = objects[(size_t)orow * OBJ_D + k];
      else if (k < 164)  v = er[(size_t)orow * EP + (k - 64)];
      aF[kt][e] = (__bf16)v;
    }
  }

  float partial[8] = {0, 0, 0, 0, 0, 0, 0, 0};
#pragma unroll 1
  for (int ct = 0; ct < 7; ++ct) {
    v8f acc0 = {}, acc1 = {};
#pragma unroll
    for (int kt = 0; kt < 6; ++kt) {
      v16bf bF = *(const v16bf*)&sW[((kt * 7 + ct) * 32 + lane) * 16];
      if (kt & 1) acc1 = wmma_bf16(aF[kt], bF, acc1);
      else        acc0 = wmma_bf16(aF[kt], bF, acc0);
    }
    int col  = ct * 16 + lm;
    float bv = sB[col];
    float w2 = (col < 100) ? ow2[col] : 0.0f;
#pragma unroll
    for (int v = 0; v < 8; ++v) {
      float h = acc0[v] + acc1[v] + bv;
      h = h > 0.0f ? h : 0.0f;
      partial[v] += h * w2;
    }
  }
#pragma unroll
  for (int v = 0; v < 8; ++v) {
    partial[v] += __shfl_xor(partial[v], 1, 32);
    partial[v] += __shfl_xor(partial[v], 2, 32);
    partial[v] += __shfl_xor(partial[v], 4, 32);
    partial[v] += __shfl_xor(partial[v], 8, 32);
  }
  if (lm == 0) {
    float b2 = ob2[0];
#pragma unroll
    for (int v = 0; v < 8; ++v) {
      float x = partial[v] + b2;
      out[r0 + v + 8 * half] = 1.0f / (1.0f + expf(-x));
    }
  }
}

// ---------------------------------------------------------------------------
// Host launcher
// ---------------------------------------------------------------------------
extern "C" void kernel_launch(void* const* d_in, const int* in_sizes, int n_in,
                              void* d_out, int out_size, void* d_ws, size_t ws_size,
                              hipStream_t stream) {
  (void)in_sizes; (void)n_in; (void)out_size; (void)ws_size;
  const float* objects = (const float*)d_in[0];
  const float* srel    = (const float*)d_in[1];
  const float* rrel    = (const float*)d_in[2];
  const float* rinfo   = (const float*)d_in[3];
  const float* rw1 = (const float*)d_in[4];
  const float* rb1 = (const float*)d_in[5];
  const float* rw2 = (const float*)d_in[6];
  const float* rb2 = (const float*)d_in[7];
  const float* rw3 = (const float*)d_in[8];
  const float* rb3 = (const float*)d_in[9];
  const float* rw4 = (const float*)d_in[10];
  const float* rb4 = (const float*)d_in[11];
  const float* ow1 = (const float*)d_in[12];
  const float* ob1 = (const float*)d_in[13];
  const float* ow2 = (const float*)d_in[14];
  const float* ob2 = (const float*)d_in[15];
  float* out = (float*)d_out;

  // Workspace carve-up (~227 MB total), 256B aligned slices.
  char* ws = (char*)d_ws;
  size_t off = 0;
  auto carve = [&](size_t bytes) -> void* {
    void* p = ws + off;
    off = (off + bytes + 255) & ~(size_t)255;
    return p;
  };
  __bf16* X    = (__bf16*)carve((size_t)MROWS * XK * 2);   // 83.2 MB
  __bf16* H    = (__bf16*)carve((size_t)MROWS * HP * 2);   // 83.2 MB
  __bf16* EFF  = (__bf16*)carve((size_t)MROWS * EP * 2);   // 58.3 MB
  float*  ER   = (float*) carve((size_t)Bn * Nn * EP * 4); // 0.9 MB
  __bf16* W1p  = (__bf16*)carve(5 * 10 * 512 * 2);
  __bf16* W2p  = (__bf16*)carve(5 * 10 * 512 * 2);
  __bf16* W3p  = (__bf16*)carve(5 * 10 * 512 * 2);
  __bf16* W4p  = (__bf16*)carve(5 * 7 * 512 * 2);
  __bf16* OW1p = (__bf16*)carve(6 * 7 * 512 * 2);
  __bf16* OBJp = (__bf16*)carve((size_t)Bn * 16 * 512 * 2);
  float*  BP   = (float*) carve(5 * 160 * 4);

  // --- prep -----------------------------------------------------------------
  pack_weight_kernel<<<(5 * 10 * 512 + 255) / 256, 256, 0, stream>>>(rw1, W1p, 160, 150, 5, 10);
  pack_weight_kernel<<<(5 * 10 * 512 + 255) / 256, 256, 0, stream>>>(rw2, W2p, 150, 150, 5, 10);
  pack_weight_kernel<<<(5 * 10 * 512 + 255) / 256, 256, 0, stream>>>(rw3, W3p, 150, 150, 5, 10);
  pack_weight_kernel<<<(5 * 7 * 512 + 255) / 256, 256, 0, stream>>>(rw4, W4p, 150, 100, 5, 7);
  pack_weight_kernel<<<(6 * 7 * 512 + 255) / 256, 256, 0, stream>>>(ow1, OW1p, 164, 100, 6, 7);
  pack_objects_kernel<<<(Bn * 16 * 512 + 255) / 256, 256, 0, stream>>>(objects, OBJp);
  pack_bias_kernel<<<(5 * 160 + 255) / 256, 256, 0, stream>>>(rb1, rb2, rb3, rb4, ob1, BP);
  zero_kernel<<<((long)Bn * Nn * EP + 255) / 256, 256, 0, stream>>>(ER, (long)Bn * Nn * EP);

  // --- gather (WMMA) --------------------------------------------------------
  gather_kernel<<<Bn * 254, 128, 0, stream>>>(srel, rrel, rinfo, OBJp, X);

  // --- relation MLP (WMMA, ping-pong X <-> H, final to EFF) -----------------
  layer_kernel<5, 10><<<2032, 256, 0, stream>>>(X, XK, H, HP, W1p, BP + 0);
  layer_kernel<5, 10><<<2032, 256, 0, stream>>>(H, HP, X, XK, W2p, BP + 160);
  layer_kernel<5, 10><<<2032, 256, 0, stream>>>(X, XK, H, HP, W3p, BP + 320);
  layer_kernel<5, 7> <<<2032, 256, 0, stream>>>(H, HP, EFF, EP, W4p, BP + 480);

  // --- scatter (WMMA + f32 atomics) ----------------------------------------
  scatter_kernel<<<(Bn * 8 * 7 * 8) / 8, 256, 0, stream>>>(rrel, EFF, ER);

  // --- object MLP + sigmoid -------------------------------------------------
  final_kernel<<<32, 128, 0, stream>>>(objects, ER, OW1p, BP + 640, ow2, ob2, out);
}
